// DecoderWithAttention_50826642980915
// MI455X (gfx1250) — matrix operand order repaired and checked by reading, not compiled, not run on hardware
//
#include <hip/hip_runtime.h>

// ---------------------------------------------------------------------------
// Problem dims (from reference)
// ---------------------------------------------------------------------------
#define BB   128
#define PATT 36
#define PF   37          // P_ATT + 1 (img_fc appended)
#define FF   2048
#define EE   1024
#define DD   1024
#define AA   1024
#define OO   512
#define VV   10000
#define LL   20
#define TT   19          // L - 1 decode steps

typedef __bf16 bf16_t;
typedef __attribute__((ext_vector_type(16))) __bf16 v16bf;
typedef __attribute__((ext_vector_type(8)))  float  v8f;

__device__ __forceinline__ float sigmoidf_(float x) { return 1.0f / (1.0f + __expf(-x)); }

// ---------------------------------------------------------------------------
// WMMA GEMM:   C[m,n] = act( A[m,:] . B[n,:] + bias[n] )
//   A: [M,K] bf16 row-major (lda) activations; B: [N,K] bf16 row-major (ldb)
//   weights (PyTorch Linear layout -> effectively A @ B^T).
//   C: f32 (Cf) and/or bf16 (Cb), row stride ldc (elements).
//   declens/t: if declens != null and t >= declens[m] -> write 0 (active mask).
//
// One wave computes a 64x16 strip: 4 M-tiles share each B (weight) fragment,
// so the L2-bandwidth-critical weight stream is read once per 4 WMMAs.
// Requires M % 64 == 0, N % 16 == 0, K % 32 == 0 (all call sites satisfy).
// ---------------------------------------------------------------------------
#define MT 4
__global__ void wmma_gemm_bf16(const bf16_t* __restrict__ A, int lda,
                               const bf16_t* __restrict__ Bm, int ldb,
                               const float* __restrict__ bias,
                               float* Cf, bf16_t* Cb, long ldc,
                               const int* __restrict__ declens, int t,
                               int relu, int M, int N, int K)
{
    const int lane = threadIdx.x & 31;
    const int wave = threadIdx.x >> 5;
    const int ntn  = N >> 4;
    const int tile = blockIdx.x * (blockDim.x >> 5) + wave;
    if (tile >= (M >> 6) * ntn) return;
    const int mblk = tile / ntn;          // 64-row block
    const int nt   = tile % ntn;

    const int l15 = lane & 15;
    const int hi  = (lane >> 4) & 1;

    const bf16_t* arow = A  + (long)((mblk << 6) + l15) * lda;   // first M-tile row
    const bf16_t* brow = Bm + (long)((nt << 4) + l15) * ldb;

    v8f acc[MT] = {};
    union U { v16bf v; unsigned int u[8]; } a, b;

    for (int kc = 0; kc < K; kc += 32) {
        if (kc + 64 <= K)                          // -> global_prefetch_b8 (weights)
            __builtin_prefetch(brow + kc + 32, 0, 1);

        // B 32x16 bf16 fragment: lanes 0-15 K=0..15, lanes 16-31 K=16..31
#pragma unroll
        for (int v = 0; v < 8; ++v) {
            int kb = kc + (v << 1) + (hi ? 16 : 0);
            b.u[v] = *(const unsigned int*)(brow + kb);
        }
        // 4 M-tiles reuse this weight fragment
#pragma unroll
        for (int mt = 0; mt < MT; ++mt) {
            const bf16_t* ar = arow + (long)(mt << 4) * lda;
#pragma unroll
            for (int v = 0; v < 8; ++v) {
                // A 16x32 bf16: lanes 0-15 K=0..15, lanes 16-31 K=16..31
                int ka = kc + (v << 1) + (hi ? 16 : 0);
                a.u[v] = *(const unsigned int*)(ar + ka);
            }
            acc[mt] = __builtin_amdgcn_wmma_f32_16x16x32_bf16(
                          false, a.v, false, b.v, (short)0, acc[mt], false, false);
        }
    }

#pragma unroll
    for (int mt = 0; mt < MT; ++mt) {
#pragma unroll
        for (int v = 0; v < 8; ++v) {
            int mr = (mblk << 6) + (mt << 4) + v + (hi ? 8 : 0); // C: VGPR v -> M
            int nc = (nt << 4) + l15;                            //    lane   -> N
            float val = acc[mt][v] + (bias ? bias[nc] : 0.0f);
            if (relu && val < 0.0f) val = 0.0f;
            if (declens && t >= declens[mr]) val = 0.0f;
            long off = (long)mr * ldc + nc;
            if (Cf) Cf[off] = val;
            if (Cb) Cb[off] = (bf16_t)val;
        }
    }
}

// ---------------------------------------------------------------------------
// One-time prep: stable descending sort by caption length (one block, 128 thr)
// ---------------------------------------------------------------------------
__global__ void sort_prep(const int* __restrict__ enc_caps,
                          const int* __restrict__ cap_len,
                          int* enc_sorted, int* declens,
                          float* out_enc, float* out_declen, float* out_sortind)
{
    __shared__ int len[BB];
    __shared__ int sidx[BB];
    int i = threadIdx.x;
    len[i] = cap_len[i];
    __syncthreads();
    int pos = 0;
    for (int j = 0; j < BB; ++j) {
        int lj = len[j];
        if (lj > len[i] || (lj == len[i] && j < i)) ++pos;
    }
    sidx[pos] = i;
    __syncthreads();
    int src = sidx[i];
    declens[i]     = len[src] - 1;
    out_declen[i]  = (float)(len[src] - 1);
    out_sortind[i] = (float)src;
    for (int k = 0; k < LL; ++k) {
        int tok = enc_caps[src * LL + k];
        enc_sorted[i * LL + k] = tok;
        out_enc[i * LL + k]    = (float)tok;
    }
}

// fp32 -> bf16 copy
__global__ void cvt_bf16(const float* __restrict__ s, bf16_t* __restrict__ d, long n)
{
    long i = (long)blockIdx.x * blockDim.x + threadIdx.x;
    if (i < n) d[i] = (bf16_t)s[i];
}

// concat two fp32 weight matrices column-wise into one bf16 matrix [rows, k1+k2]
__global__ void concat_w(const float* __restrict__ w1, int k1,
                         const float* __restrict__ w2, int k2,
                         bf16_t* __restrict__ dst, int rows)
{
    int kk = k1 + k2;
    long n = (long)rows * kk;
    long i = (long)blockIdx.x * blockDim.x + threadIdx.x;
    if (i >= n) return;
    int r = (int)(i / kk), c = (int)(i % kk);
    float v = (c < k1) ? w1[(long)r * k1 + c] : w2[(long)r * k2 + (c - k1)];
    dst[i] = (bf16_t)v;
}

// gather sorted image features [B,37,F] bf16 (p==36 row = img_fc), and img_fc bf16
__global__ void gather_imgfeat(const float* __restrict__ img_att,
                               const float* __restrict__ img_fc,
                               const float* __restrict__ sortind_f,
                               bf16_t* __restrict__ imgfeat,
                               bf16_t* __restrict__ imgfcb)
{
    long i = (long)blockIdx.x * blockDim.x + threadIdx.x;
    if (i >= (long)BB * PF * FF) return;
    int b   = (int)(i / (PF * FF));
    int rem = (int)(i % (PF * FF));
    int p = rem / FF, f = rem % FF;
    int sb = (int)sortind_f[b];
    float v = (p < PATT) ? img_att[((long)sb * PATT + p) * FF + f]
                         : img_fc[(long)sb * FF + f];
    imgfeat[i] = (bf16_t)v;
    if (p == PATT) imgfcb[(long)b * FF + f] = (bf16_t)v;
}

// zero decoder state
__global__ void zero_state(float* h1, float* c1, float* h2, float* c2,
                           bf16_t* h1b, bf16_t* h2b)
{
    int i = blockIdx.x * blockDim.x + threadIdx.x;
    if (i >= BB * DD) return;
    h1[i] = c1[i] = h2[i] = c2[i] = 0.0f;
    h1b[i] = (bf16_t)0.0f;
    h2b[i] = (bf16_t)0.0f;
}

// build X_ll [B,5120] = [emb_t(1024) | h2(1024) | img_fc(2048) | h1(1024)] bf16
__global__ void build_xll(bf16_t* __restrict__ X,
                          const float* __restrict__ emb,
                          const int* __restrict__ enc_sorted, int t,
                          const bf16_t* __restrict__ h1b,
                          const bf16_t* __restrict__ h2b,
                          const bf16_t* __restrict__ imgfcb)
{
    int i = blockIdx.x * blockDim.x + threadIdx.x;
    const int KX = EE + DD + FF + DD;   // 5120
    if (i >= BB * KX) return;
    int b = i / KX, c = i % KX;
    bf16_t val;
    if (c < EE) {
        int tok = enc_sorted[b * LL + t];
        val = (bf16_t)emb[(long)tok * EE + c];
    } else if (c < EE + DD) {
        val = h2b[b * DD + (c - EE)];
    } else if (c < EE + DD + FF) {
        val = imgfcb[b * FF + (c - EE - DD)];
    } else {
        val = h1b[b * DD + (c - EE - DD - FF)];
    }
    X[i] = val;
}

// LSTM cell pointwise: gates [B,4096] (i|f|g|o) -> h_new/c_new; conditional commit
__global__ void lstm_point(const float* __restrict__ gates,
                           float* __restrict__ h, float* __restrict__ c,
                           bf16_t* __restrict__ hb_commit,
                           bf16_t* __restrict__ hnb,
                           const int* __restrict__ declens, int t)
{
    int i = blockIdx.x * blockDim.x + threadIdx.x;
    if (i >= BB * DD) return;
    int b = i >> 10, j = i & (DD - 1);
    const float* g = gates + (long)b * 4 * DD;
    float ig = sigmoidf_(g[j]);
    float fg = sigmoidf_(g[DD + j]);
    float gg = tanhf(g[2 * DD + j]);
    float og = sigmoidf_(g[3 * DD + j]);
    float cn = fg * c[i] + ig * gg;
    float hn = og * tanhf(cn);
    hnb[i] = (bf16_t)hn;                 // h_new always feeds downstream this step
    if (t < declens[b]) {                // commit carry only for active rows
        c[i] = cn;
        h[i] = hn;
        hb_commit[i] = (bf16_t)hn;
    }
}

// scores[b,p] = sum_a relu(att1[b,p,a] + d1[b,a] + d2[b,a]) * wfull[a] + bfull
// one wave per (b,p)
__global__ void att_scores(const float* __restrict__ att1,
                           const float* __restrict__ d1,
                           const float* __restrict__ d2,
                           const float* __restrict__ wfull,
                           const float* __restrict__ bfull,
                           float* __restrict__ scores)
{
    int lane = threadIdx.x & 31;
    int w = blockIdx.x * (blockDim.x >> 5) + (threadIdx.x >> 5);
    if (w >= BB * PF) return;
    int b = w / PF;
    const float* e  = att1 + (long)w * AA;
    const float* b1 = d1 + (long)b * AA;
    const float* b2 = d2 + (long)b * AA;
    float s = 0.0f;
    for (int a = lane; a < AA; a += 32) {
        float v = e[a] + b1[a] + b2[a];
        v = v > 0.0f ? v : 0.0f;
        s += v * wfull[a];
    }
    for (int off = 16; off; off >>= 1) s += __shfl_xor(s, off, 32);
    if (lane == 0) scores[w] = s + bfull[0];
}

// softmax over p + awe pooling; writes alphas (masked) and X_al = [awe | h2]
__global__ void softmax_awe(const float* __restrict__ scores,
                            const bf16_t* __restrict__ imgfeat,
                            bf16_t* __restrict__ Xal,
                            const bf16_t* __restrict__ h2b,
                            float* __restrict__ alphas_out,
                            const int* __restrict__ declens, int t)
{
    __shared__ float al[PF];
    int b = blockIdx.x, tid = threadIdx.x;
    if (tid == 0) {
        float m = -1e30f;
        for (int p = 0; p < PF; ++p) { float v = scores[b * PF + p]; al[p] = v; if (v > m) m = v; }
        float s = 0.0f;
        for (int p = 0; p < PF; ++p) { al[p] = __expf(al[p] - m); s += al[p]; }
        float inv = 1.0f / s;
        for (int p = 0; p < PF; ++p) al[p] *= inv;
    }
    __syncthreads();
    int active = (t < declens[b]) ? 1 : 0;
    if (tid < PF) alphas_out[((long)b * TT + t) * PF + tid] = active ? al[tid] : 0.0f;
    const int KX = FF + DD;   // 3072
    for (int f = tid; f < FF; f += blockDim.x) {
        float s = 0.0f;
        for (int p = 0; p < PF; ++p)
            s += al[p] * (float)imgfeat[((long)b * PF + p) * FF + f];
        Xal[(long)b * KX + f] = (bf16_t)s;
    }
    for (int j = tid; j < DD; j += blockDim.x)
        Xal[(long)b * KX + FF + j] = h2b[b * DD + j];
}

// ---------------------------------------------------------------------------
// Host driver
// ---------------------------------------------------------------------------
extern "C" void kernel_launch(void* const* d_in, const int* in_sizes, int n_in,
                              void* d_out, int out_size, void* d_ws, size_t ws_size,
                              hipStream_t stream)
{
    (void)in_sizes; (void)n_in; (void)out_size; (void)ws_size;

    // inputs (setup_inputs order)
    const float* img_att = (const float*)d_in[0];
    const float* img_fc  = (const float*)d_in[1];
    const int*   enc     = (const int*)d_in[2];
    const int*   caplen  = (const int*)d_in[3];
    const float* emb     = (const float*)d_in[4];
    const float* Wfa     = (const float*)d_in[5];
    const float* bfa     = (const float*)d_in[6];
    const float* Wd1     = (const float*)d_in[7];
    const float* bd1     = (const float*)d_in[8];
    const float* Wd2     = (const float*)d_in[9];
    const float* bd2     = (const float*)d_in[10];
    const float* Wfull   = (const float*)d_in[11];
    const float* bfull   = (const float*)d_in[12];
    const float* Wlang   = (const float*)d_in[13];
    const float* blang   = (const float*)d_in[14];
    const float* Watt    = (const float*)d_in[15];
    const float* batt    = (const float*)d_in[16];
    const float* Wout    = (const float*)d_in[17];
    const float* bout    = (const float*)d_in[18];
    const float* Wll_ih  = (const float*)d_in[19];
    const float* Wll_hh  = (const float*)d_in[20];
    const float* bll     = (const float*)d_in[21];
    const float* Wal_ih  = (const float*)d_in[22];
    const float* Wal_hh  = (const float*)d_in[23];
    const float* bal     = (const float*)d_in[24];
    const float* Wfc     = (const float*)d_in[25];
    const float* bfc     = (const float*)d_in[26];

    float* out = (float*)d_out;
    const long PRED  = 0;
    const long P1    = (long)BB * TT * VV;
    const long ENCO  = 2 * P1;
    const long DECLO = ENCO + (long)BB * LL;
    const long ALPH  = DECLO + BB;
    const long SIND  = ALPH + (long)BB * TT * PF;

    // workspace bump allocator (256B aligned)
    char* ws = (char*)d_ws;
    size_t off = 0;
    auto alloc = [&](size_t bytes) {
        void* p = ws + off;
        off = (off + bytes + 255) & ~(size_t)255;
        return p;
    };
    const int KLL = EE + DD + FF + DD;   // 5120
    const int KAL = FF + DD;             // 3072

    bf16_t* Wll_cat = (bf16_t*)alloc((size_t)4 * DD * KLL * 2);
    bf16_t* Wal_cat = (bf16_t*)alloc((size_t)4 * DD * KAL * 2);
    bf16_t* Wfc_b   = (bf16_t*)alloc((size_t)VV * DD * 2);
    bf16_t* Wout_b  = (bf16_t*)alloc((size_t)VV * 2 * OO * 2);
    bf16_t* Wfa_b   = (bf16_t*)alloc((size_t)AA * FF * 2);
    bf16_t* Wd1_b   = (bf16_t*)alloc((size_t)AA * DD * 2);
    bf16_t* Wd2_b   = (bf16_t*)alloc((size_t)AA * DD * 2);
    bf16_t* Wlang_b = (bf16_t*)alloc((size_t)OO * DD * 2);
    bf16_t* Watt_b  = (bf16_t*)alloc((size_t)OO * DD * 2);
    bf16_t* imgfeat = (bf16_t*)alloc((size_t)BB * PF * FF * 2);
    bf16_t* imgfcb  = (bf16_t*)alloc((size_t)BB * FF * 2);
    float*  att1    = (float*)alloc((size_t)BB * PF * AA * 4);
    bf16_t* Xll     = (bf16_t*)alloc((size_t)BB * KLL * 2);
    bf16_t* Xal     = (bf16_t*)alloc((size_t)BB * KAL * 2);
    float*  gates   = (float*)alloc((size_t)BB * 4 * DD * 4);
    float*  h1      = (float*)alloc((size_t)BB * DD * 4);
    float*  c1      = (float*)alloc((size_t)BB * DD * 4);
    float*  h2      = (float*)alloc((size_t)BB * DD * 4);
    float*  c2      = (float*)alloc((size_t)BB * DD * 4);
    bf16_t* h1b     = (bf16_t*)alloc((size_t)BB * DD * 2);
    bf16_t* h2b     = (bf16_t*)alloc((size_t)BB * DD * 2);
    bf16_t* h1nb    = (bf16_t*)alloc((size_t)BB * DD * 2);
    bf16_t* h2nb    = (bf16_t*)alloc((size_t)BB * DD * 2);
    float*  d1buf   = (float*)alloc((size_t)BB * AA * 4);
    float*  d2buf   = (float*)alloc((size_t)BB * AA * 4);
    float*  scores  = (float*)alloc((size_t)BB * PF * 4);
    bf16_t* hid_b   = (bf16_t*)alloc((size_t)BB * 2 * OO * 2);
    int*    enc_s   = (int*)alloc((size_t)BB * LL * 4);
    int*    declens = (int*)alloc((size_t)BB * 4);

    auto gemm = [&](const bf16_t* A, int lda, const bf16_t* B, int ldb,
                    const float* bias, float* Cf, bf16_t* Cb, long ldc,
                    const int* mask, int t, int relu, int M, int N, int K) {
        int tiles = (M >> 6) * (N >> 4);            // 64x16 strips
        int grid = (tiles + 3) / 4;                 // 4 waves / block
        wmma_gemm_bf16<<<grid, 128, 0, stream>>>(A, lda, B, ldb, bias,
                                                 Cf, Cb, ldc, mask, t, relu, M, N, K);
    };
    auto nb = [](long n) { return (int)((n + 255) / 256); };

    // ---- one-time prep ----
    sort_prep<<<1, BB, 0, stream>>>(enc, caplen, enc_s, declens,
                                    out + ENCO, out + DECLO, out + SIND);
    concat_w<<<nb((long)4 * DD * KLL), 256, 0, stream>>>(Wll_ih, EE + DD + FF, Wll_hh, DD,
                                                         Wll_cat, 4 * DD);
    concat_w<<<nb((long)4 * DD * KAL), 256, 0, stream>>>(Wal_ih, FF, Wal_hh, DD,
                                                         Wal_cat, 4 * DD);
    cvt_bf16<<<nb((long)VV * DD), 256, 0, stream>>>(Wfc,  Wfc_b,  (long)VV * DD);
    cvt_bf16<<<nb((long)VV * 2 * OO), 256, 0, stream>>>(Wout, Wout_b, (long)VV * 2 * OO);
    cvt_bf16<<<nb((long)AA * FF), 256, 0, stream>>>(Wfa,  Wfa_b,  (long)AA * FF);
    cvt_bf16<<<nb((long)AA * DD), 256, 0, stream>>>(Wd1,  Wd1_b,  (long)AA * DD);
    cvt_bf16<<<nb((long)AA * DD), 256, 0, stream>>>(Wd2,  Wd2_b,  (long)AA * DD);
    cvt_bf16<<<nb((long)OO * DD), 256, 0, stream>>>(Wlang, Wlang_b, (long)OO * DD);
    cvt_bf16<<<nb((long)OO * DD), 256, 0, stream>>>(Watt,  Watt_b,  (long)OO * DD);
    gather_imgfeat<<<nb((long)BB * PF * FF), 256, 0, stream>>>(img_att, img_fc,
                                                               out + SIND, imgfeat, imgfcb);
    zero_state<<<nb((long)BB * DD), 256, 0, stream>>>(h1, c1, h2, c2, h1b, h2b);

    // att1 = imgfeat @ Wfa^T + bfa   [4736,1024]
    gemm(imgfeat, FF, Wfa_b, FF, bfa, att1, nullptr, AA,
         nullptr, 0, 0, BB * PF, AA, FF);

    // ---- decode loop ----
    for (int t = 0; t < TT; ++t) {
        build_xll<<<nb((long)BB * KLL), 256, 0, stream>>>(Xll, emb, enc_s, t,
                                                          h1b, h2b, imgfcb);
        // LSTM1 gates
        gemm(Xll, KLL, Wll_cat, KLL, bll, gates, nullptr, 4 * DD,
             nullptr, 0, 0, BB, 4 * DD, KLL);
        lstm_point<<<nb((long)BB * DD), 256, 0, stream>>>(gates, h1, c1, h1b, h1nb,
                                                          declens, t);
        // preds1 -> d_out[P1 + b*T*V + t*V + :]
        gemm(h1nb, DD, Wfc_b, DD, bfc, out + P1 + (long)t * VV, nullptr, (long)TT * VV,
             declens, t, 0, BB, VV, DD);
        // attention projections
        gemm(h1nb, DD, Wd1_b, DD, bd1, d1buf, nullptr, AA, nullptr, 0, 0, BB, AA, DD);
        gemm(h2b,  DD, Wd2_b, DD, bd2, d2buf, nullptr, AA, nullptr, 0, 0, BB, AA, DD);
        att_scores<<<(BB * PF + 3) / 4, 128, 0, stream>>>(att1, d1buf, d2buf,
                                                          Wfull, bfull, scores);
        softmax_awe<<<BB, 256, 0, stream>>>(scores, imgfeat, Xal, h2b,
                                            out + ALPH, declens, t);
        // LSTM2 gates
        gemm(Xal, KAL, Wal_cat, KAL, bal, gates, nullptr, 4 * DD,
             nullptr, 0, 0, BB, 4 * DD, KAL);
        lstm_point<<<nb((long)BB * DD), 256, 0, stream>>>(gates, h2, c2, h2b, h2nb,
                                                          declens, t);
        // hid = relu([h1n@Wlang^T+blang | h2n@Watt^T+batt])  (bf16)
        gemm(h1nb, DD, Wlang_b, DD, blang, nullptr, hid_b,       2 * OO,
             nullptr, 0, 1, BB, OO, DD);
        gemm(h2nb, DD, Watt_b,  DD, batt,  nullptr, hid_b + OO,  2 * OO,
             nullptr, 0, 1, BB, OO, DD);
        // out -> d_out[PRED + b*T*V + t*V + :]
        gemm(hid_b, 2 * OO, Wout_b, 2 * OO, bout, out + PRED + (long)t * VV, nullptr,
             (long)TT * VV, declens, t, 0, BB, VV, 2 * OO);
    }
}
